// MultiHeadAttention_74474732912811
// MI455X (gfx1250) — matrix operand order repaired and checked by reading, not compiled, not run on hardware
//
#include <hip/hip_runtime.h>

typedef __attribute__((ext_vector_type(16))) _Float16 v16h;
typedef __attribute__((ext_vector_type(8)))  _Float16 v8h;
typedef __attribute__((ext_vector_type(8)))  float    v8f;

constexpr int B_  = 4;
constexpr int T_  = 2048;
constexpr int D_  = 1024;
constexpr int H_  = 16;
constexpr int DHd = 64;
constexpr int TCV = 682;        // conv output length
constexpr int TC  = 683;        // compressed length (1 + conv)
constexpr int TCP = 704;        // padded Tc (multiple of 32, aligned loads)

// ---------------- WMMA helpers (layouts per CDNA5 ISA 7.12.2) ----------------

__device__ __forceinline__ v8f wmma_f16(v16h a, v16h b, v8f c) {
    return __builtin_amdgcn_wmma_f32_16x16x32_f16(false, a, false, b,
                                                  (short)0, c, false, false);
}

// A fragment: 16x32 f16 tile at (row0, k0) from row-major src (lda halves).
// lane L: m = L&15, half = L>>4; K = {8*half+e | e<8} U {16+8*half+(e-8)}
__device__ __forceinline__ v16h load_a_frag(const _Float16* __restrict__ src,
                                            int row0, int maxrow, int lda,
                                            int k0, int lane) {
    int m = lane & 15, half = lane >> 4;
    int r = row0 + m; if (r >= maxrow) r = maxrow - 1;
    const _Float16* p = src + (size_t)r * lda + k0 + 8 * half;
    v8h lo = *(const v8h*)p;
    v8h hi = *(const v8h*)(p + 16);
    return __builtin_shufflevector(lo, hi, 0,1,2,3,4,5,6,7,8,9,10,11,12,13,14,15);
}

// B fragment for C = A * W^T, W row-major [N, K] (ldw halves).
// lane L: n = L&15, K = {8*half+e | e<8} U {16+8*half+(e-8)}
__device__ __forceinline__ v16h load_bT_frag(const _Float16* __restrict__ w,
                                             int n0, int maxn, int ldw,
                                             int k0, int lane) {
    int n = n0 + (lane & 15); if (n >= maxn) n = maxn - 1;
    int half = lane >> 4;
    const _Float16* p = w + (size_t)n * ldw + k0 + 8 * half;
    v8h lo = *(const v8h*)p;
    v8h hi = *(const v8h*)(p + 16);
    return __builtin_shufflevector(lo, hi, 0,1,2,3,4,5,6,7,8,9,10,11,12,13,14,15);
}

// ---------------- small prep kernels ----------------

__global__ void cvt_f32_f16(const float* __restrict__ src, _Float16* __restrict__ dst, int n) {
    int i = blockIdx.x * blockDim.x + threadIdx.x;
    if (i < n) dst[i] = (_Float16)src[i];
}

// conv_w (OIH: [D][D][3]) -> 3 contiguous [D][D] f16 matrices (per tap h)
__global__ void pack_convw(const float* __restrict__ cw, _Float16* __restrict__ out) {
    int i = blockIdx.x * blockDim.x + threadIdx.x;
    if (i < 3 * D_ * D_) {
        int h = i / (D_ * D_);
        int rem = i - h * D_ * D_;
        int o = rem / D_, ii = rem - o * D_;
        out[i] = (_Float16)cw[((size_t)o * D_ + ii) * 3 + h];
    }
}

// k_c row 0 per batch = original k row 0
__global__ void kc_row0(const _Float16* __restrict__ kh, _Float16* __restrict__ kc) {
    int i = blockIdx.x * blockDim.x + threadIdx.x;
    if (i < B_ * D_) {
        int b = i >> 10, d = i & (D_ - 1);
        kc[(size_t)b * TC * D_ + d] = kh[(size_t)b * T_ * D_ + d];
    }
}

// ---------------- conv as WMMA GEMM: kc[1+p] = sum_h K[3p+h] @ Wc[h]^T ----------------

__device__ __forceinline__ v16h load_a_conv(const _Float16* __restrict__ kb,
                                            int p0, int h, int k0, int lane) {
    int m = lane & 15, half = lane >> 4;
    int p = p0 + m; if (p > TCV - 1) p = TCV - 1;
    const _Float16* ptr = kb + (size_t)(3 * p + h) * D_ + k0 + 8 * half;
    v8h lo = *(const v8h*)ptr;
    v8h hi = *(const v8h*)(ptr + 16);
    return __builtin_shufflevector(lo, hi, 0,1,2,3,4,5,6,7,8,9,10,11,12,13,14,15);
}

__global__ void __launch_bounds__(256)
conv_gemm(const _Float16* __restrict__ kh, const _Float16* __restrict__ wc,
          _Float16* __restrict__ kc) {
    const int MT = 43, NT = 16; // 43 m-tiles (682 rows), 16 n-tiles of 64
    int gw = (blockIdx.x * blockDim.x + threadIdx.x) >> 5;
    int lane = threadIdx.x & 31;
    if (gw >= B_ * MT * NT) return;
    int b = gw / (MT * NT);
    int rem = gw - b * MT * NT;
    int mt = rem / NT, nt = rem - mt * NT;
    int row0 = mt * 16, col0 = nt * 64;
    const _Float16* kb = kh + (size_t)b * T_ * D_;

    v8f acc[4] = {};
    for (int h = 0; h < 3; h++) {
        const _Float16* w = wc + (size_t)h * D_ * D_;
        for (int k0 = 0; k0 < D_; k0 += 32) {
            v16h a = load_a_conv(kb, row0, h, k0, lane);
#pragma unroll
            for (int t = 0; t < 4; t++) {
                v16h bb = load_bT_frag(w, col0 + 16 * t, D_, D_, k0, lane);
                acc[t] = wmma_f16(a, bb, acc[t]);
            }
        }
    }
    int half = lane >> 4, nl = lane & 15;
#pragma unroll
    for (int t = 0; t < 4; t++)
#pragma unroll
        for (int r = 0; r < 8; r++) {
            int p = row0 + r + 8 * half;
            if (p < TCV)
                kc[((size_t)b * TC + 1 + p) * D_ + col0 + 16 * t + nl] = (_Float16)acc[t][r];
        }
}

// ---------------- generic projection GEMM: Y = X @ W^T, scatter epilogue ----------------
// mode 0: qp  [B,H,T,64]      mode 1: kp [B,H,TCP,64]
// mode 2: vpT [B,H,64,TCP]    mode 3: fp32 out [M,D] + bias
__global__ void __launch_bounds__(256)
proj_gemm(const _Float16* __restrict__ X, const _Float16* __restrict__ W,
          void* __restrict__ outp, const float* __restrict__ bias,
          int M, int rows_per_b, int mode) {
    int gw = (blockIdx.x * blockDim.x + threadIdx.x) >> 5;
    int lane = threadIdx.x & 31;
    int mtiles = (M + 15) >> 4;
    if (gw >= mtiles * 16) return;
    int mt = gw >> 4, nt = gw & 15;
    int row0 = mt * 16, col0 = nt * 64;

    v8f acc[4] = {};
    for (int k0 = 0; k0 < D_; k0 += 32) {
        v16h a = load_a_frag(X, row0, M, D_, k0, lane);
#pragma unroll
        for (int t = 0; t < 4; t++) {
            v16h bb = load_bT_frag(W, col0 + 16 * t, D_, D_, k0, lane);
            acc[t] = wmma_f16(a, bb, acc[t]);
        }
    }
    int half = lane >> 4, nl = lane & 15;
#pragma unroll
    for (int t = 0; t < 4; t++)
#pragma unroll
        for (int r = 0; r < 8; r++) {
            int m = row0 + r + 8 * half;
            if (m >= M) continue;
            int ng = col0 + 16 * t + nl;
            int b = m / rows_per_b, loc = m - b * rows_per_b;
            int h = ng >> 6, dh = ng & 63;
            float v = acc[t][r];
            if (mode == 0)
                ((_Float16*)outp)[(((size_t)b * H_ + h) * T_ + loc) * DHd + dh] = (_Float16)v;
            else if (mode == 1)
                ((_Float16*)outp)[(((size_t)b * H_ + h) * TCP + loc) * DHd + dh] = (_Float16)v;
            else if (mode == 2)
                ((_Float16*)outp)[(((size_t)b * H_ + h) * DHd + dh) * TCP + loc] = (_Float16)v;
            else
                ((float*)outp)[(size_t)m * D_ + ng] = v + bias[ng];
        }
}

// ---------------- flash attention: one wave per 16-query tile ----------------

__global__ void __launch_bounds__(256)
attn_kernel(const _Float16* __restrict__ qp, const _Float16* __restrict__ kp,
            const _Float16* __restrict__ vpT, _Float16* __restrict__ o) {
    __shared__ __align__(16) _Float16 pbuf[8][16 * 32];
    int wib = threadIdx.x >> 5, lane = threadIdx.x & 31;
    int gw = blockIdx.x * 8 + wib;
    if (gw >= B_ * H_ * (T_ / 16)) return;
    int bh = gw >> 7;            // T_/16 = 128 q-tiles
    int qt = gw & 127;
    int i0 = qt << 4;
    int b = bh >> 4, h = bh & 15;

    const _Float16* Q = qp + (size_t)bh * T_ * DHd;
    const _Float16* K = kp + (size_t)bh * TCP * DHd;
    const _Float16* V = vpT + (size_t)bh * DHd * TCP;
    int half = lane >> 4, nl = lane & 15;

    v16h a0 = load_a_frag(Q, i0, T_, DHd, 0, lane);
    v16h a1 = load_a_frag(Q, i0, T_, DHd, 32, lane);

    v8f oacc[4] = {};
    float rowM[8], rowL[8];
#pragma unroll
    for (int r = 0; r < 8; r++) { rowM[r] = -3.0e38f; rowL[r] = 0.f; }

    int jend = (i0 + 15) / 3 + 1;           // causal: allowed iff 3j <= i
    if (jend > TC) jend = TC;

    for (int j0 = 0; j0 < jend; j0 += 32) {
        v8f s0 = {}, s1 = {};
        {
            v16h bk = load_bT_frag(K, j0, TCP, DHd, 0, lane);
            s0 = wmma_f16(a0, bk, s0);
            bk = load_bT_frag(K, j0, TCP, DHd, 32, lane);
            s0 = wmma_f16(a1, bk, s0);
            bk = load_bT_frag(K, j0 + 16, TCP, DHd, 0, lane);
            s1 = wmma_f16(a0, bk, s1);
            bk = load_bT_frag(K, j0 + 16, TCP, DHd, 32, lane);
            s1 = wmma_f16(a1, bk, s1);
        }
        float sv0[8], sv1[8];
#pragma unroll
        for (int r = 0; r < 8; r++) {
            int i = i0 + r + 8 * half;
            int ja = j0 + nl, jb = j0 + 16 + nl;
            sv0[r] = (3 * ja > i) ? -1e9f : s0[r] * 0.125f;
            sv1[r] = (3 * jb > i) ? -1e9f : s1[r] * 0.125f;
        }
#pragma unroll
        for (int r = 0; r < 8; r++) {
            float tmax = fmaxf(sv0[r], sv1[r]);
#pragma unroll
            for (int off = 1; off < 16; off <<= 1)
                tmax = fmaxf(tmax, __shfl_xor(tmax, off, 16));
            float mnew = fmaxf(rowM[r], tmax);
            float scale = __expf(rowM[r] - mnew);
            rowM[r] = mnew;
            float p0 = __expf(sv0[r] - mnew);
            float p1 = __expf(sv1[r] - mnew);
            sv0[r] = p0; sv1[r] = p1;
            float ps = p0 + p1;
#pragma unroll
            for (int off = 1; off < 16; off <<= 1)
                ps += __shfl_xor(ps, off, 16);
            rowL[r] = rowL[r] * scale + ps;
#pragma unroll
            for (int t = 0; t < 4; t++) oacc[t][r] *= scale;
        }
        // C-layout P -> LDS -> A-layout fragment
#pragma unroll
        for (int r = 0; r < 8; r++) {
            pbuf[wib][(r + 8 * half) * 32 + nl]      = (_Float16)sv0[r];
            pbuf[wib][(r + 8 * half) * 32 + 16 + nl] = (_Float16)sv1[r];
        }
        asm volatile("s_wait_dscnt 0x0" ::: "memory");
        v16h pa;
        {
            const _Float16* pp = &pbuf[wib][nl * 32 + 8 * half];
            v8h lo = *(const v8h*)pp;
            v8h hi = *(const v8h*)(pp + 16);
            pa = __builtin_shufflevector(lo, hi, 0,1,2,3,4,5,6,7,8,9,10,11,12,13,14,15);
        }
#pragma unroll
        for (int t = 0; t < 4; t++) {
            v16h vb = load_bT_frag(V, 16 * t, DHd, TCP, j0, lane);
            oacc[t] = wmma_f16(pa, vb, oacc[t]);
        }
    }
#pragma unroll
    for (int t = 0; t < 4; t++)
#pragma unroll
        for (int r = 0; r < 8; r++) {
            int i = i0 + r + 8 * half;
            float val = oacc[t][r] / rowL[r];
            o[((size_t)b * T_ + i) * D_ + h * DHd + 16 * t + nl] = (_Float16)val;
        }
}

// ---------------- host launch ----------------

extern "C" void kernel_launch(void* const* d_in, const int* in_sizes, int n_in,
                              void* d_out, int out_size, void* d_ws, size_t ws_size,
                              hipStream_t stream) {
    (void)in_sizes; (void)n_in; (void)out_size; (void)ws_size;
    const float* q  = (const float*)d_in[0];
    const float* k  = (const float*)d_in[1];
    // d_in[2] = v : unused by reference ('c' mode replaces V with compressed K)
    const float* Wq = (const float*)d_in[3];
    const float* Wk = (const float*)d_in[4];
    const float* Wv = (const float*)d_in[5];
    const float* Wo = (const float*)d_in[6];
    const float* bo = (const float*)d_in[7];
    const float* cw = (const float*)d_in[8];

    char* ws = (char*)d_ws;
    size_t off = 0;
    auto alloc = [&](size_t elems) -> _Float16* {
        _Float16* p = (_Float16*)(ws + off);
        off += (elems * sizeof(_Float16) + 255) & ~(size_t)255;
        return p;
    };
    _Float16* q_h   = alloc((size_t)B_ * T_ * D_);
    _Float16* k_h   = alloc((size_t)B_ * T_ * D_);
    _Float16* Wq_h  = alloc((size_t)D_ * D_);
    _Float16* Wk_h  = alloc((size_t)D_ * D_);
    _Float16* Wv_h  = alloc((size_t)D_ * D_);
    _Float16* Wo_h  = alloc((size_t)D_ * D_);
    _Float16* Wc_h  = alloc((size_t)3 * D_ * D_);
    _Float16* kc_h  = alloc((size_t)B_ * TC * D_);
    _Float16* qp_h  = alloc((size_t)B_ * H_ * T_ * DHd);
    _Float16* kp_h  = alloc((size_t)B_ * H_ * TCP * DHd);
    _Float16* vpT_h = alloc((size_t)B_ * H_ * DHd * TCP);
    _Float16* o_h   = alloc((size_t)B_ * T_ * D_);

    int n;
    n = B_ * T_ * D_;
    cvt_f32_f16<<<(n + 255) / 256, 256, 0, stream>>>(q, q_h, n);
    cvt_f32_f16<<<(n + 255) / 256, 256, 0, stream>>>(k, k_h, n);
    n = D_ * D_;
    cvt_f32_f16<<<(n + 255) / 256, 256, 0, stream>>>(Wq, Wq_h, n);
    cvt_f32_f16<<<(n + 255) / 256, 256, 0, stream>>>(Wk, Wk_h, n);
    cvt_f32_f16<<<(n + 255) / 256, 256, 0, stream>>>(Wv, Wv_h, n);
    cvt_f32_f16<<<(n + 255) / 256, 256, 0, stream>>>(Wo, Wo_h, n);
    n = 3 * D_ * D_;
    pack_convw<<<(n + 255) / 256, 256, 0, stream>>>(cw, Wc_h);
    n = B_ * D_;
    kc_row0<<<(n + 255) / 256, 256, 0, stream>>>(k_h, kc_h);

    // conv GEMM: 4 * 43 * 16 waves
    conv_gemm<<<(B_ * 43 * 16) / 8, 256, 0, stream>>>(k_h, Wc_h, kc_h);

    // projections
    {
        int M = B_ * T_;                       // 8192
        int blocks = (((M + 15) / 16) * 16 * 32 + 255) / 256;
        proj_gemm<<<blocks, 256, 0, stream>>>(q_h, Wq_h, qp_h, nullptr, M, T_, 0);
    }
    {
        int M = B_ * TC;                       // 2732
        int blocks = (((M + 15) / 16) * 16 * 32 + 255) / 256;
        proj_gemm<<<blocks, 256, 0, stream>>>(kc_h, Wk_h, kp_h, nullptr, M, TC, 1);
        proj_gemm<<<blocks, 256, 0, stream>>>(kc_h, Wv_h, vpT_h, nullptr, M, TC, 2);
    }

    // flash attention: B*H * (T/16) waves = 8192
    attn_kernel<<<(B_ * H_ * (T_ / 16)) / 8, 256, 0, stream>>>(qp_h, kp_h, vpT_h, o_h);

    // output projection + bias -> fp32 d_out
    {
        int M = B_ * T_;
        int blocks = (((M + 15) / 16) * 16 * 32 + 255) / 256;
        proj_gemm<<<blocks, 256, 0, stream>>>(o_h, Wo_h, d_out, bo, M, T_, 3);
    }
}
